// sgconv_66228395705231
// MI455X (gfx1250) — compile-verified, matrix operand order checked by reading, not run on hardware
//
#include <hip/hip_runtime.h>
#include <hip/hip_bf16.h>
#include <math.h>

// ---------------------------------------------------------------------------
// SGC (K=3) on gfx1250:
//   deg -> dinv -> 3x (self-term + edge scatter-add) -> WMMA classifier
//                   -> relu -> log_softmax
// ---------------------------------------------------------------------------

typedef __attribute__((ext_vector_type(2)))  float    v2f;
typedef __attribute__((ext_vector_type(8)))  float    v8f;
typedef __attribute__((ext_vector_type(16))) _Float16 v16h;

#if defined(__has_builtin)
#  if __has_builtin(__builtin_amdgcn_wmma_f32_16x16x4_f32)
#    define USE_WMMA_F32 1
#  endif
#endif
#ifndef USE_WMMA_F32
#  define USE_WMMA_F32 0
#endif

#define NFEAT 64

// --------------------------- degree / norm ---------------------------------

__global__ void k_zero_u32(unsigned* __restrict__ p, int n) {
    int i = blockIdx.x * blockDim.x + threadIdx.x;
    if (i < n) p[i] = 0u;
}

__global__ void k_count_deg(const long long* __restrict__ ei,
                            unsigned* __restrict__ deg, int n_edges) {
    int e = blockIdx.x * blockDim.x + threadIdx.x;
    if (e < n_edges) {
        int c = (int)ei[(size_t)n_edges + e];   // target node
        atomicAdd(&deg[c], 1u);
    }
}

__global__ void k_dinv(const unsigned* __restrict__ deg,
                       float* __restrict__ dinv, int n) {
    int i = blockIdx.x * blockDim.x + threadIdx.x;
    if (i < n) dinv[i] = rsqrtf((float)(deg[i] + 1u));  // +1 self loop
}

// --------------------------- propagation -----------------------------------

// out[i] = dinv[i]^2 * in[i]   (self-loop message), float4 per thread
__global__ void k_hop_self(const float* __restrict__ hin, float* __restrict__ hout,
                           const float* __restrict__ dinv, int n_nodes) {
    int t = blockIdx.x * blockDim.x + threadIdx.x;
    if (t >= n_nodes * 16) return;
    int i  = t >> 4;
    int f4 = (t & 15) << 2;
    float d = dinv[i];
    float s = d * d;
    const float4 v = *(const float4*)(hin + (size_t)i * NFEAT + f4);
    float4 o;
    o.x = s * v.x; o.y = s * v.y; o.z = s * v.z; o.w = s * v.w;
    *(float4*)(hout + (size_t)i * NFEAT + f4) = o;
}

// out[col] += dinv[row]*dinv[col] * in[row];  16 threads (4x float4) per edge
__global__ void k_hop_scatter(const float* __restrict__ hin, float* __restrict__ hout,
                              const long long* __restrict__ ei,
                              const float* __restrict__ dinv, int n_edges) {
    long long t = (long long)blockIdx.x * blockDim.x + threadIdx.x;
    if (t >= (long long)n_edges * 16) return;
    int e  = (int)(t >> 4);
    int f4 = ((int)t & 15) << 2;
    int r = (int)ei[e];                       // source
    int c = (int)ei[(size_t)n_edges + e];     // target
    float nrm = dinv[r] * dinv[c];
    const float4 v = *(const float4*)(hin + (size_t)r * NFEAT + f4);
    float* o = hout + (size_t)c * NFEAT + f4;
    atomicAdd(o + 0, nrm * v.x);
    atomicAdd(o + 1, nrm * v.y);
    atomicAdd(o + 2, nrm * v.z);
    atomicAdd(o + 3, nrm * v.w);
}

// ---------------------- classifier + log-softmax ---------------------------
// One wave per 16-node tile.  D[16x48] = A[16x64] * B[64x48] via WMMA,
// weights zero-padded 40->48 classes and staged in LDS.

__global__ void __launch_bounds__(256) k_classify(
    const float* __restrict__ h, const float* __restrict__ W,
    const float* __restrict__ bias, float* __restrict__ out,
    int n_nodes, int n_classes)
{
    __shared__ float sW[48 * NFEAT];        // zero-padded W [48][64]
    __shared__ float sb[48];                // zero-padded bias
    __shared__ float sm[8][16][48];         // per-wave logits tile

    const int tid = threadIdx.x;

    // cooperative staging (all waves participate before any early exit)
    for (int i = tid; i < 48 * NFEAT; i += 256) {
        int c = i >> 6;
        sW[i] = (c < n_classes) ? W[i] : 0.0f;
    }
    if (tid < 48) sb[tid] = (tid < n_classes) ? bias[tid] : 0.0f;
    __syncthreads();

    const int lane   = tid & 31;
    const int wv     = tid >> 5;
    const int ntiles = (n_nodes + 15) >> 4;
    const int tile   = blockIdx.x * 8 + wv;
    if (tile >= ntiles) return;             // wave-uniform: EXEC stays full
    const int node0 = tile << 4;

    const int p  = lane & 15;               // row (A) / col (B) within tile
    const int hi = lane >> 4;               // K-half selector

    int arow = node0 + p;
    if (arow >= n_nodes) arow = n_nodes - 1;
    const float* __restrict__ A = h + (size_t)arow * NFEAT;

    v8f acc0 = {}, acc1 = {}, acc2 = {};

#if USE_WMMA_F32
    // exact fp32 path: V_WMMA_F32_16X16X4_F32, 16 K-steps
    for (int kk = 0; kk < NFEAT; kk += 4) {
        const int k0 = kk + 2 * hi;         // lanes 16..31 hold K+2,K+3
        v2f a, b0, b1, b2;
        a.x = A[k0];  a.y = A[k0 + 1];
        const float* w0 = &sW[(p)      * NFEAT + k0];
        const float* w1 = &sW[(16 + p) * NFEAT + k0];
        const float* w2 = &sW[(32 + p) * NFEAT + k0];
        b0.x = w0[0]; b0.y = w0[1];
        b1.x = w1[0]; b1.y = w1[1];
        b2.x = w2[0]; b2.y = w2[1];
        acc0 = __builtin_amdgcn_wmma_f32_16x16x4_f32(false, a, false, b0, (short)0, acc0, false, false);
        acc1 = __builtin_amdgcn_wmma_f32_16x16x4_f32(false, a, false, b1, (short)0, acc1, false, false);
        acc2 = __builtin_amdgcn_wmma_f32_16x16x4_f32(false, a, false, b2, (short)0, acc2, false, false);
    }
#else
    // fallback: f16 inputs, f32 accumulate (codegen-confirmed builtin)
    for (int kk = 0; kk < NFEAT; kk += 32) {
        v16h a, b0, b1, b2;
        for (int hh = 0; hh < 16; ++hh) {
            int K = hh + ((hh >= 8) ? 8 : 0) + (hi ? 8 : 0);  // ISA 16-bit A layout
            a[hh]  = (_Float16)A[kk + K];
            b0[hh] = (_Float16)sW[(p)      * NFEAT + kk + K];
            b1[hh] = (_Float16)sW[(16 + p) * NFEAT + kk + K];
            b2[hh] = (_Float16)sW[(32 + p) * NFEAT + kk + K];
        }
        acc0 = __builtin_amdgcn_wmma_f32_16x16x32_f16(false, a, false, b0, (short)0, acc0, false, false);
        acc1 = __builtin_amdgcn_wmma_f32_16x16x32_f16(false, a, false, b1, (short)0, acc1, false, false);
        acc2 = __builtin_amdgcn_wmma_f32_16x16x32_f16(false, a, false, b2, (short)0, acc2, false, false);
    }
#endif

    // bias + relu into this wave's LDS tile
    for (int v = 0; v < 8; ++v) {
        const int m = v + hi * 8;           // C/D layout: lanes 16..31 -> M+8
        sm[wv][m][p]      = fmaxf(acc0[v] + sb[p],      0.0f);
        sm[wv][m][16 + p] = fmaxf(acc1[v] + sb[16 + p], 0.0f);
        sm[wv][m][32 + p] = fmaxf(acc2[v] + sb[32 + p], 0.0f);
    }
    asm volatile("s_wait_dscnt 0" ::: "memory");   // wave-local LDS RAW fence

    // per-row log_softmax, lanes 0..15 each own one row
    if (lane < 16) {
        const int m    = lane;
        const int node = node0 + m;
        if (node < n_nodes) {
            float mx = 0.0f;                         // relu'd, so >= 0
            for (int c = 0; c < n_classes; ++c) mx = fmaxf(mx, sm[wv][m][c]);
            float se = 0.0f;
            for (int c = 0; c < n_classes; ++c) se += expf(sm[wv][m][c] - mx);
            const float lse = logf(se) + mx;
            float* __restrict__ o = out + (size_t)node * n_classes;
            for (int c = 0; c < n_classes; ++c) o[c] = sm[wv][m][c] - lse;
        }
    }
}

// ------------------------------ launcher -----------------------------------

extern "C" void kernel_launch(void* const* d_in, const int* in_sizes, int n_in,
                              void* d_out, int out_size, void* d_ws, size_t ws_size,
                              hipStream_t stream) {
    const float*     x  = (const float*)d_in[0];
    const long long* ei = (const long long*)d_in[1];   // int64 edge_index [2,E]
    const float*     W  = (const float*)d_in[2];
    const float*     b  = (const float*)d_in[3];
    float*           out = (float*)d_out;

    const int N = in_sizes[0] / NFEAT;   // 100000
    const int E = in_sizes[1] / 2;       // 1250000
    const int C = in_sizes[3];           // 40

    // workspace layout: dinv[N] | deg[N] | hA[N*64] | hB[N*64]  (~52 MB)
    float*    dinv = (float*)d_ws;
    unsigned* deg  = (unsigned*)d_ws + N;
    float*    hA   = (float*)d_ws + 2 * (size_t)N;
    float*    hB   = hA + (size_t)N * NFEAT;

    const int B256 = 256;

    k_zero_u32 <<<(N + 255) / 256, B256, 0, stream>>>(deg, N);
    k_count_deg<<<(E + 255) / 256, B256, 0, stream>>>(ei, deg, E);
    k_dinv     <<<(N + 255) / 256, B256, 0, stream>>>(deg, dinv, N);

    const float* src = x;
    float* dst = hA;
    for (int hop = 0; hop < 3; ++hop) {
        long long nt = (long long)N * 16;
        k_hop_self<<<(int)((nt + 255) / 256), B256, 0, stream>>>(src, dst, dinv, N);
        long long et = (long long)E * 16;
        k_hop_scatter<<<(int)((et + 255) / 256), B256, 0, stream>>>(src, dst, ei, dinv, E);
        src = dst;
        dst = (dst == hA) ? hB : hA;
    }
    // src now holds h after 3 hops (hA)

    const int ntiles = (N + 15) / 16;        // 6250
    const int blocks = (ntiles + 7) / 8;     // 8 waves per block
    k_classify<<<blocks, 256, 0, stream>>>(src, W, b, out, N, C);
}
